// MoEFFN_7181185319146
// MI455X (gfx1250) — matrix operand order repaired and checked by reading, not compiled, not run on hardware
//
#include <hip/hip_runtime.h>

// ---------------------------------------------------------------------------
// MoE FFN forward for MI455X (gfx1250, wave32, WMMA f32_16x16x32_f16,
// async global->LDS double-buffered staging)
// ---------------------------------------------------------------------------
#define NTOK   4096     // B*T
#define DMODEL 512
#define DHID   1024
#define NEXP   64
#define TOPK   8
#define CAP    1024

#define BM 128          // M tile (8 waves x 16 rows)
#define BN 64           // N tile (4 WMMA col-tiles)
#define BK 32           // K step (one WMMA K)
#define LDSS 40         // padded LDS stride in halves (80B -> conflict-free b128)

typedef __attribute__((ext_vector_type(16))) _Float16 v16h;
typedef __attribute__((ext_vector_type(8)))  _Float16 v8h;
typedef __attribute__((ext_vector_type(8)))  float    v8f;

typedef int v4i __attribute__((vector_size(16)));
typedef __attribute__((address_space(1))) v4i glob_v4i;   // global 16B chunk
typedef __attribute__((address_space(3))) v4i lds_v4i;    // LDS 16B chunk

#if defined(__has_builtin)
#  if __has_builtin(__builtin_amdgcn_global_load_async_to_lds_b128)
#    define USE_ASYNC_LDS 1
#  endif
#endif
#ifndef USE_ASYNC_LDS
#  define USE_ASYNC_LDS 0
#endif

// 16-byte global -> LDS copy (async on CDNA5 when available)
__device__ __forceinline__ void copy16_g2l(const _Float16* g, _Float16* l) {
#if USE_ASYNC_LDS
    __builtin_amdgcn_global_load_async_to_lds_b128(
        (glob_v4i*)(size_t)g,
        (lds_v4i*)(unsigned)(size_t)l, 0, 0);
#else
    *(v8h*)l = *(const v8h*)g;
#endif
}

__device__ __forceinline__ void wait_async_stage() {
#if USE_ASYNC_LDS
    asm volatile("s_wait_asynccnt 0" ::: "memory");
#endif
}

// ------------------------- fp32 -> f16 converts ----------------------------
__global__ void cvt_f16_kernel(const float* __restrict__ in, _Float16* __restrict__ out, long n) {
    long i = (long)blockIdx.x * blockDim.x + threadIdx.x;
    long stride = (long)gridDim.x * blockDim.x;
    for (; i < n; i += stride) out[i] = (_Float16)in[i];
}

// batched: out[e, c, r] = in[e, r, c]  (expert weights -> [N,K] output-major)
__global__ void transpose_cvt_batched_kernel(const float* __restrict__ in,
                                             _Float16* __restrict__ out,
                                             int R, int C, long nTotal) {
    long rc = (long)R * C;
    long stride = (long)gridDim.x * blockDim.x;
    for (long i = (long)blockIdx.x * blockDim.x + threadIdx.x; i < nTotal; i += stride) {
        long e = i / rc;
        long rem = i - e * rc;
        int r = (int)(rem / C), c = (int)(rem % C);
        out[e * rc + (long)c * R + r] = (_Float16)in[i];
    }
}

// ------------------------- templated WMMA GEMM -----------------------------
// C[M,Nld] (+)= A[M,K] * B^T where B is [Nld,K] row-major (output-major).
// DUAL   : two B matrices, out = f16( silu(A*B1) * (A*B2) )
// GATHER : A rows gathered through slot_tok (per-expert capacity buffer)
// SCATTER: epilogue atomically adds w*row into token rows of fp32 out
template<bool DUAL, bool GATHER, bool SCATTER>
__global__ void __launch_bounds__(256)
gemm_wmma_kernel(const _Float16* __restrict__ A,
                 const _Float16* __restrict__ B1,
                 const _Float16* __restrict__ B2,
                 void* __restrict__ Out,
                 int M, int K, int Nld,
                 long eStrideA, long eStrideB, long eStrideOut,
                 const int*   __restrict__ slot_tok,
                 const float* __restrict__ slot_w,
                 const int*   __restrict__ counts)
{
    constexpr bool EXPERT = GATHER || SCATTER;

    __shared__ __align__(16) _Float16 lA[2 * BM * LDSS];
    __shared__ __align__(16) _Float16 lB1[2 * BN * LDSS];
    __shared__ __align__(16) _Float16 lB2[DUAL ? 2 * BN * LDSS : 16];

    const int tid  = threadIdx.x;
    const int wave = tid >> 5;
    const int lane = tid & 31;
    const int rowBase = blockIdx.x * BM;
    const int nBase   = blockIdx.y * BN;
    const int ex      = EXPERT ? (int)blockIdx.z : 0;

    int cnt = M;
    if constexpr (EXPERT) {
        cnt = counts[ex];
        if (cnt > M) cnt = M;
        if (rowBase >= cnt) return;              // skip empty capacity tiles
    }

    const _Float16* Ae  = A  + (long)ex * eStrideA;
    const _Float16* B1e = B1 + (long)ex * eStrideB;
    const _Float16* B2e = DUAL ? (B2 + (long)ex * eStrideB) : (const _Float16*)nullptr;
    const int* st = EXPERT ? (slot_tok + (long)ex * CAP) : (const int*)nullptr;

    // stage one [BM x BK] A tile and [BN x BK] B tile(s) into LDS buffer `buf`
    auto stage = [&](int k0, int buf) {
        _Float16* dA = lA + buf * (BM * LDSS);
#pragma unroll
        for (int it = 0; it < 2; ++it) {
            int idx = tid + it * 256;            // 512 chunks of 8 halves
            int row = idx >> 2;
            int seg = idx & 3;
            int gr  = rowBase + row;
            const _Float16* src;
            if constexpr (GATHER) {
                int tok = 0;                      // invalid rows -> slot 0 (discarded later)
                if (gr < cnt) tok = st[gr];
                src = Ae + (long)tok * K + k0 + seg * 8;
            } else {
                src = Ae + (long)gr * K + k0 + seg * 8;
            }
            copy16_g2l(src, dA + row * LDSS + seg * 8);
        }
        {
            int row = tid >> 2;                  // 0..63 output rows of B
            int seg = tid & 3;
            _Float16* dB = lB1 + buf * (BN * LDSS);
            copy16_g2l(B1e + (long)(nBase + row) * K + k0 + seg * 8,
                       dB + row * LDSS + seg * 8);
            if constexpr (DUAL) {
                _Float16* dB2 = lB2 + buf * (BN * LDSS);
                copy16_g2l(B2e + (long)(nBase + row) * K + k0 + seg * 8,
                           dB2 + row * LDSS + seg * 8);
            }
        }
    };

    v8f acc1[4];
    v8f acc2[DUAL ? 4 : 1];
#pragma unroll
    for (int n = 0; n < 4; ++n)
#pragma unroll
        for (int j = 0; j < 8; ++j) {
            acc1[n][j] = 0.0f;
            if constexpr (DUAL) acc2[n][j] = 0.0f;
        }

    // ---- software-pipelined main loop (double-buffered LDS) ----
    stage(0, 0);
    wait_async_stage();
    __syncthreads();

    int cur = 0;
    for (int k0 = 0; k0 < K; k0 += BK) {
        int nxt = cur ^ 1;
        if (k0 + BK < K) stage(k0 + BK, nxt);    // prefetch next tile (async)

        const _Float16* cA  = lA  + cur * (BM * LDSS);
        const _Float16* cB1 = lB1 + cur * (BN * LDSS);
        const _Float16* cB2 = lB2 + cur * (BN * LDSS);

        // fragments (A 16x32 / B 32x16, contiguous k = 16*(lane>>4) + s for both A and B)
        const int mrow = wave * 16 + (lane & 15);
        const int kA   = (lane >> 4) * 16;       // lanes16-31 hold K=16..31
        v8h alo = *(const v8h*)(cA + mrow * LDSS + kA);        // K = kA..kA+7
        v8h ahi = *(const v8h*)(cA + mrow * LDSS + kA + 8);    // K = kA+8..kA+15
        v16h af;
#pragma unroll
        for (int q = 0; q < 8; ++q) { af[q] = alo[q]; af[q + 8] = ahi[q]; }

        const int kB = (lane >> 4) * 16;         // lanes16-31 hold K=16..31
#pragma unroll
        for (int n = 0; n < 4; ++n) {
            int col = n * 16 + (lane & 15);
            v8h blo = *(const v8h*)(cB1 + col * LDSS + kB);
            v8h bhi = *(const v8h*)(cB1 + col * LDSS + kB + 8);
            v16h bf;
#pragma unroll
            for (int q = 0; q < 8; ++q) { bf[q] = blo[q]; bf[q + 8] = bhi[q]; }
            acc1[n] = __builtin_amdgcn_wmma_f32_16x16x32_f16(
                false, af, false, bf, (short)0, acc1[n], false, false);
            if constexpr (DUAL) {
                v8h clo = *(const v8h*)(cB2 + col * LDSS + kB);
                v8h chi = *(const v8h*)(cB2 + col * LDSS + kB + 8);
                v16h cf;
#pragma unroll
                for (int q = 0; q < 8; ++q) { cf[q] = clo[q]; cf[q + 8] = chi[q]; }
                acc2[n] = __builtin_amdgcn_wmma_f32_16x16x32_f16(
                    false, af, false, cf, (short)0, acc2[n], false, false);
            }
        }
        wait_async_stage();                      // next tile resident in LDS
        __syncthreads();                         // + everyone done reading cur
        cur = nxt;
    }

    // ---- epilogue (C layout: VGPR j -> row j + 8*(lane>>4), col = lane&15) ----
    float*    outF = (float*)Out;
    _Float16* outH = (_Float16*)Out;
#pragma unroll
    for (int n = 0; n < 4; ++n) {
#pragma unroll
        for (int j = 0; j < 8; ++j) {
            int rloc = wave * 16 + (lane >> 4) * 8 + j;
            int gr   = rowBase + rloc;
            int gc   = nBase + n * 16 + (lane & 15);
            float v  = acc1[n][j];
            if constexpr (DUAL) {
                float g = v;
                float u = acc2[n][j];
                float hval = (g / (1.0f + __expf(-g))) * u;   // silu(g)*u
                if (!EXPERT || gr < cnt)
                    outH[(long)ex * eStrideOut + (long)gr * Nld + gc] = (_Float16)hval;
            } else if constexpr (SCATTER) {
                if (gr < cnt) {
                    int   tok = st[gr];
                    float wgt = slot_w[(long)ex * CAP + gr];
                    atomicAdd(outF + (long)tok * Nld + gc, wgt * v);
                }
            } else {
                outF[(long)gr * Nld + gc] = v;
            }
        }
    }
}

// ---------------- router: softmax over 128 logits + top-8 ------------------
// one wave per token; lane l owns logits {l, l+32 (real), l+64, l+96 (null)}
__global__ void router_topk_kernel(const float* __restrict__ logits,      // [N,64]
                                   const float* __restrict__ logit_bias,  // [64]
                                   const float* __restrict__ null_logit_p,// [1]
                                   int*   __restrict__ counts,            // [64]
                                   int*   __restrict__ slot_tok,          // [64,CAP]
                                   float* __restrict__ slot_w,            // [64,CAP]
                                   float* __restrict__ P_sum,             // [64]
                                   float* __restrict__ aux_acc,           // [0]=sum lse^2, [1]=null count
                                   int N)
{
    const int lane = threadIdx.x & 31;
    const int wave = threadIdx.x >> 5;
    const int t = blockIdx.x * (blockDim.x >> 5) + wave;
    if (t >= N) return;

    const float nl = null_logit_p[0];
    float l0 = logits[(long)t * 64 + lane]      + logit_bias[lane];
    float l1 = logits[(long)t * 64 + 32 + lane] + logit_bias[32 + lane];

    // softmax over all 128
    float m = fmaxf(fmaxf(l0, l1), nl);
#pragma unroll
    for (int s = 16; s > 0; s >>= 1) m = fmaxf(m, __shfl_xor(m, s, 32));
    float e0 = __expf(l0 - m), e1 = __expf(l1 - m), en = __expf(nl - m);
    float dsum = e0 + e1 + en + en;
#pragma unroll
    for (int s = 16; s > 0; s >>= 1) dsum += __shfl_xor(dsum, s, 32);
    float inv = 1.0f / dsum;

    float p[4]     = { e0 * inv, e1 * inv, en * inv, en * inv };
    int   id[4]    = { lane, lane + 32, lane + 64, lane + 96 };
    bool  alive[4] = { true, true, true, true };

    float wsum = 0.0f;
    int   my_e = -1;          // pick owned by lane k (k<8)
    float my_w = 0.0f;
    int   nnull = 0;
    for (int k = 0; k < TOPK; ++k) {
        float bv = -1.0f; int bi = 0x7fffffff;
#pragma unroll
        for (int q = 0; q < 4; ++q)
            if (alive[q] && (p[q] > bv || (p[q] == bv && id[q] < bi))) { bv = p[q]; bi = id[q]; }
#pragma unroll
        for (int s = 16; s > 0; s >>= 1) {
            float ov = __shfl_xor(bv, s, 32);
            int   oi = __shfl_xor(bi, s, 32);
            if (ov > bv || (ov == bv && oi < bi)) { bv = ov; bi = oi; }
        }
#pragma unroll
        for (int q = 0; q < 4; ++q) if (id[q] == bi) alive[q] = false;
        if (bi < NEXP) wsum += bv; else ++nnull;
        if (lane == k) { my_e = bi; my_w = bv; }
    }
    float invw = 1.0f / fmaxf(wsum, 1e-6f);

    // slot assignment: lanes 0..7 each own one pick
    if (lane < TOPK && my_e >= 0 && my_e < NEXP) {
        int pos = atomicAdd(&counts[my_e], 1);
        if (pos < CAP) {
            slot_tok[(long)my_e * CAP + pos] = t;
            slot_w[(long)my_e * CAP + pos]  = my_w * invw;
        }
    }

    // P_real: softmax over the 64 real logits only
    float mr = fmaxf(l0, l1);
#pragma unroll
    for (int s = 16; s > 0; s >>= 1) mr = fmaxf(mr, __shfl_xor(mr, s, 32));
    float r0 = __expf(l0 - mr), r1 = __expf(l1 - mr);
    float dr = r0 + r1;
#pragma unroll
    for (int s = 16; s > 0; s >>= 1) dr += __shfl_xor(dr, s, 32);
    float invr = 1.0f / dr;
    atomicAdd(&P_sum[lane],      r0 * invr);
    atomicAdd(&P_sum[lane + 32], r1 * invr);

    if (lane == 0) {
        float lse = m + __logf(dsum);
        atomicAdd(&aux_acc[0], lse * lse);
        atomicAdd(&aux_acc[1], (float)nnull);
    }
}

// ---------------------------- aux loss scalar ------------------------------
__global__ void finalize_aux_kernel(const int* __restrict__ counts,
                                    const float* __restrict__ P_sum,
                                    const float* __restrict__ aux_acc,
                                    float* __restrict__ out_aux, int N)
{
    __shared__ float s1[64], s2[64];
    int e = threadIdx.x;
    float c = (float)counts[e];
    s1[e] = c;
    __syncthreads();
    for (int s = 32; s > 0; s >>= 1) { if (e < s) s1[e] += s1[e + s]; __syncthreads(); }
    float total = s1[0];
    __syncthreads();
    float f = c / fmaxf(total, 1e-6f);
    float P = P_sum[e] / (float)N;
    s2[e] = f * P;
    __syncthreads();
    for (int s = 32; s > 0; s >>= 1) { if (e < s) s2[e] += s2[e + s]; __syncthreads(); }
    if (e == 0) {
        float L_bal = (float)NEXP * s2[0];
        float L_z   = aux_acc[0] / (float)N;
        float null_rate = aux_acc[1] / (float)(N * TOPK);
        float L_null = (null_rate - 0.5f) * (null_rate - 0.5f);
        out_aux[0] = 0.02f * L_bal + 0.001f * L_z + 0.01f * L_null;
    }
}

// ---------------------------------------------------------------------------
extern "C" void kernel_launch(void* const* d_in, const int* in_sizes, int n_in,
                              void* d_out, int out_size, void* d_ws, size_t ws_size,
                              hipStream_t stream)
{
    (void)in_sizes; (void)n_in; (void)out_size; (void)ws_size;

    const float* x          = (const float*)d_in[0];
    const float* gate_w     = (const float*)d_in[1];
    const float* logit_bias = (const float*)d_in[2];
    const float* null_logit = (const float*)d_in[3];
    const float* W_gate     = (const float*)d_in[4];
    const float* W_up       = (const float*)d_in[5];
    const float* W_down     = (const float*)d_in[6];
    const float* Wsg        = (const float*)d_in[7];
    const float* Wsu        = (const float*)d_in[8];
    const float* Wsd        = (const float*)d_in[9];
    float* out = (float*)d_out;

    // ---- carve workspace ----
    char* ws = (char*)d_ws;
    size_t off = 0;
    auto alloc = [&](size_t bytes) -> char* {
        char* p = ws + off;
        off = (off + bytes + 255) & ~(size_t)255;
        return p;
    };
    _Float16* xh       = (_Float16*)alloc((size_t)NTOK * DMODEL * 2);
    _Float16* gw_h     = (_Float16*)alloc((size_t)NEXP * DMODEL * 2);              // [64,512]  (N,K)
    _Float16* Wsg_h    = (_Float16*)alloc((size_t)DHID * DMODEL * 2);              // [1024,512]
    _Float16* Wsu_h    = (_Float16*)alloc((size_t)DHID * DMODEL * 2);
    _Float16* Wsd_h    = (_Float16*)alloc((size_t)DMODEL * DHID * 2);              // [512,1024]
    _Float16* WgT_h    = (_Float16*)alloc((size_t)NEXP * DHID * DMODEL * 2);       // [E,1024,512]
    _Float16* WuT_h    = (_Float16*)alloc((size_t)NEXP * DHID * DMODEL * 2);
    _Float16* WdT_h    = (_Float16*)alloc((size_t)NEXP * DMODEL * DHID * 2);       // [E,512,1024]
    float*    logitsF  = (float*)   alloc((size_t)NTOK * NEXP * 4);
    _Float16* shared_h = (_Float16*)alloc((size_t)NTOK * DHID * 2);
    _Float16* hbuf     = (_Float16*)alloc((size_t)NEXP * CAP * DHID * 2);
    int*      slot_tok = (int*)     alloc((size_t)NEXP * CAP * 4);
    float*    slot_w   = (float*)   alloc((size_t)NEXP * CAP * 4);
    char*     ctrs     = alloc(1024);
    int*      counts   = (int*)ctrs;
    float*    P_sum    = (float*)(ctrs + 64 * 4);
    float*    aux_acc  = (float*)(ctrs + 128 * 4);

    dim3 blk(256);

    // ---- precision converts (f32 -> f16, all B matrices as [N,K]) ----
    cvt_f16_kernel<<<2048, blk, 0, stream>>>(x,      xh,    (long)NTOK * DMODEL);
    cvt_f16_kernel<<<64,   blk, 0, stream>>>(gate_w, gw_h,  (long)NEXP * DMODEL);
    cvt_f16_kernel<<<1024, blk, 0, stream>>>(Wsg,    Wsg_h, (long)DHID * DMODEL);
    cvt_f16_kernel<<<1024, blk, 0, stream>>>(Wsu,    Wsu_h, (long)DHID * DMODEL);
    cvt_f16_kernel<<<1024, blk, 0, stream>>>(Wsd,    Wsd_h, (long)DMODEL * DHID);
    transpose_cvt_batched_kernel<<<8192, blk, 0, stream>>>(
        W_gate, WgT_h, DMODEL, DHID, (long)NEXP * DMODEL * DHID);
    transpose_cvt_batched_kernel<<<8192, blk, 0, stream>>>(
        W_up,   WuT_h, DMODEL, DHID, (long)NEXP * DMODEL * DHID);
    transpose_cvt_batched_kernel<<<8192, blk, 0, stream>>>(
        W_down, WdT_h, DHID, DMODEL, (long)NEXP * DHID * DMODEL);
    (void)hipMemsetAsync(ctrs, 0, 1024, stream);

    // ---- router GEMM: logits = xh @ gate_w^T  [4096,64] ----
    gemm_wmma_kernel<false, false, false><<<dim3(NTOK / BM, NEXP / BN, 1), blk, 0, stream>>>(
        xh, gw_h, nullptr, logitsF, NTOK, DMODEL, NEXP, 0, 0, 0, nullptr, nullptr, nullptr);

    // ---- softmax / top-8 / dispatch / aux accumulation ----
    router_topk_kernel<<<NTOK / 8, blk, 0, stream>>>(
        logitsF, logit_bias, null_logit, counts, slot_tok, slot_w, P_sum, aux_acc, NTOK);

    // ---- shared expert: h = silu(x Wsg^T) * (x Wsu^T); out = h Wsd^T ----
    gemm_wmma_kernel<true, false, false><<<dim3(NTOK / BM, DHID / BN, 1), blk, 0, stream>>>(
        xh, Wsg_h, Wsu_h, shared_h, NTOK, DMODEL, DHID, 0, 0, 0, nullptr, nullptr, nullptr);
    gemm_wmma_kernel<false, false, false><<<dim3(NTOK / BM, DMODEL / BN, 1), blk, 0, stream>>>(
        shared_h, Wsd_h, nullptr, out, NTOK, DHID, DMODEL, 0, 0, 0, nullptr, nullptr, nullptr);

    // ---- routed experts: gathered dual GEMM + silu-mul -> hbuf ----
    gemm_wmma_kernel<true, true, false><<<dim3(CAP / BM, DHID / BN, NEXP), blk, 0, stream>>>(
        xh, WgT_h, WuT_h, hbuf, CAP, DMODEL, DHID,
        0, (long)DMODEL * DHID, (long)CAP * DHID, slot_tok, slot_w, counts);

    // ---- down projection + weighted scatter-add into d_out ----
    gemm_wmma_kernel<false, false, true><<<dim3(CAP / BM, DMODEL / BN, NEXP), blk, 0, stream>>>(
        hbuf, WdT_h, nullptr, out, CAP, DHID, DMODEL,
        (long)CAP * DHID, (long)DHID * DMODEL, 0, slot_tok, slot_w, counts);

    // ---- aux loss scalar ----
    finalize_aux_kernel<<<1, 64, 0, stream>>>(counts, P_sum, aux_acc,
                                              out + (long)NTOK * DMODEL, NTOK);
}
